// SelfAttentionHead_20229295964971
// MI455X (gfx1250) — matrix operand order repaired and checked by reading, not compiled, not run on hardware
//
#include <hip/hip_runtime.h>
#include <hip/hip_bf16.h>

typedef __bf16 bf16_t;
typedef __attribute__((ext_vector_type(16))) __bf16 v16bf;
typedef __attribute__((ext_vector_type(8)))  __bf16 v8bf;
typedef __attribute__((ext_vector_type(8)))  float  v8f;

#define WMMA_BF16(A, B, C) \
  __builtin_amdgcn_wmma_f32_16x16x32_bf16(false, (A), false, (B), (short)0, (C), false, false)

// Async DMA: global -> LDS, 16B per lane, tracked by ASYNCcnt (CDNA5).
// GV mode: per-lane 64-bit global address in VGPR pair, LDS byte offset in VDST VGPR.
static __device__ __forceinline__ void async_ld_b128(unsigned lds_off, const void* gaddr) {
  asm volatile("global_load_async_to_lds_b128 %0, %1, off"
               :: "v"(lds_off), "v"(gaddr) : "memory");
}
static __device__ __forceinline__ void wait_async0() {
  asm volatile("s_wait_asynccnt 0" ::: "memory");
}
static __device__ __forceinline__ unsigned lds_off_of(const void* p) {
  // flat cast of an LDS pointer: low 32 bits are the wave-relative LDS byte offset
  return (unsigned)(uintptr_t)p;
}

static __device__ __forceinline__ v16bf cat8(v8bf lo, v8bf hi) {
  v16bf r;
#pragma unroll
  for (int i = 0; i < 8; ++i) { r[i] = lo[i]; r[i + 8] = hi[i]; }
  return r;
}

// A-matrix fragment (16Mx32K, bf16), source row-major with K contiguous.
// Lane L: row = L%16; K elems {0..7,16..23} (L<16) or {8..15,24..31} (L>=16).
static __device__ __forceinline__ v16bf load_a_frag(const bf16_t* base, int stride, int lane) {
  int row  = lane & 15;
  int koff = (lane & 16) ? 8 : 0;
  const bf16_t* p = base + row * stride + koff;
  v8bf lo = *(const v8bf*)(p);
  v8bf hi = *(const v8bf*)(p + 16);
  return cat8(lo, hi);
}

// B-matrix fragment (32Kx16N, bf16), source = B^T row-major (row = N, K contiguous).
// Lane L: col = L%16; K elems {0..7,16..23} (L<16) or {8..15,24..31} (L>=16).
static __device__ __forceinline__ v16bf load_b_frag(const bf16_t* base, int stride, int lane) {
  int row  = lane & 15;
  int koff = (lane & 16) ? 8 : 0;
  const bf16_t* p = base + row * stride + koff;
  v8bf lo = *(const v8bf*)(p);
  v8bf hi = *(const v8bf*)(p + 16);
  return cat8(lo, hi);
}

// ---------------------------------------------------------------- convert ---
__global__ void cvt_bf16_kernel(const float* __restrict__ in, bf16_t* __restrict__ out, int n) {
  int i = blockIdx.x * blockDim.x + threadIdx.x;
  int stride = gridDim.x * blockDim.x;
  for (; i < n; i += stride) out[i] = (bf16_t)in[i];
}

// ------------------------------------------------------------ QKV GEMM ------
// qkv[m,o] = sum_k x[m,k]*W[o,k] + b[o];  M=8192, N=3072, K=1024.
// Double-buffered LDS tiles filled by async DMA; output scattered to Q/K/V
// in [B,H,T,hs] bf16 layout.
__global__ __launch_bounds__(256) void qkv_gemm_kernel(
    const bf16_t* __restrict__ X, const bf16_t* __restrict__ Wt,
    const float* __restrict__ bias,
    bf16_t* __restrict__ Q, bf16_t* __restrict__ K, bf16_t* __restrict__ V) {
  const int C = 1024, T = 2048, H = 16, HS = 64;
  const int LSTR = 48;  // padded LDS row stride (bf16): 96B, conflict-free b128
  const unsigned BUFB = 128 * LSTR * sizeof(bf16_t);  // 12288 B per buffer
  __shared__ bf16_t As[2][128 * LSTR];
  __shared__ bf16_t Bs[2][128 * LSTR];

  const int tid  = threadIdx.x;
  const int lane = tid & 31;
  const int w    = tid >> 5;
  const int wm   = w & 1;   // 2 m-groups of 64
  const int wn   = w >> 1;  // 4 n-groups of 32
  const int m0   = blockIdx.y * 128;
  const int n0   = blockIdx.x * 128;

  v8f acc[4][2];
#pragma unroll
  for (int mt = 0; mt < 4; ++mt)
#pragma unroll
    for (int nt = 0; nt < 2; ++nt) acc[mt][nt] = (v8f){};

  const int srow = tid >> 1, sseg = tid & 1;  // 128 rows x 2 segments of 16
  const bf16_t* xg = X  + (size_t)(m0 + srow) * C + sseg * 16;
  const bf16_t* wg = Wt + (size_t)(n0 + srow) * C + sseg * 16;
  const unsigned aoff = lds_off_of(&As[0][srow * LSTR + sseg * 16]);
  const unsigned boff = lds_off_of(&Bs[0][srow * LSTR + sseg * 16]);

  // prologue: DMA tile 0 into buffer 0
  async_ld_b128(aoff,      xg);
  async_ld_b128(aoff + 16, xg + 8);
  async_ld_b128(boff,      wg);
  async_ld_b128(boff + 16, wg + 8);

  const int NIT = C / 32;
  for (int it = 0; it < NIT; ++it) {
    wait_async0();
    __syncthreads();
    const int buf = it & 1;
    if (it + 1 < NIT) {  // DMA next tile into the other buffer (uniform branch)
      const int k1 = (it + 1) * 32;
      const unsigned bo = (buf ^ 1) * BUFB;
      async_ld_b128(aoff + bo,      xg + k1);
      async_ld_b128(aoff + bo + 16, xg + k1 + 8);
      async_ld_b128(boff + bo,      wg + k1);
      async_ld_b128(boff + bo + 16, wg + k1 + 8);
    }

    v16bf bfrag[2];
#pragma unroll
    for (int nt = 0; nt < 2; ++nt)
      bfrag[nt] = load_b_frag(&Bs[buf][(wn * 32 + nt * 16) * LSTR], LSTR, lane);
#pragma unroll
    for (int mt = 0; mt < 4; ++mt) {
      v16bf afrag = load_a_frag(&As[buf][(wm * 64 + mt * 16) * LSTR], LSTR, lane);
#pragma unroll
      for (int nt = 0; nt < 2; ++nt)
        acc[mt][nt] = WMMA_BF16(afrag, bfrag[nt], acc[mt][nt]);
    }
  }

  // Epilogue: C/D layout m=(lane/16)*8+r, n=lane%16.
  const int g = lane >> 4, n = lane & 15;
  const int b = m0 >> 11;  // m0/T, uniform (128 | 2048)
#pragma unroll
  for (int nt = 0; nt < 2; ++nt) {
    const int o    = n0 + wn * 32 + nt * 16 + n;  // output feature
    const float bv = bias[o];
    const int s    = o >> 10;        // 0=q 1=k 2=v (uniform per tile)
    const int h    = (o >> 6) & 15;  // head (uniform per tile)
    const int d    = o & 63;
    bf16_t* dst = (s == 0) ? Q : ((s == 1) ? K : V);
    dst += ((size_t)(b * H + h) * T) * HS + d;
#pragma unroll
    for (int mt = 0; mt < 4; ++mt)
#pragma unroll
      for (int r = 0; r < 8; ++r) {
        int t = (m0 + wm * 64 + mt * 16 + g * 8 + r) & (T - 1);
        dst[(size_t)t * HS] = (bf16_t)(acc[mt][nt][r] + bv);
      }
  }
}

// ----------------------------------------------------------- attention ------
// Flash attention: 1 block = 128 query rows of one (b,h); 8 waves x 16 rows.
__global__ __launch_bounds__(256) void attn_kernel(
    const bf16_t* __restrict__ Q, const bf16_t* __restrict__ Kg,
    const bf16_t* __restrict__ Vg, float* __restrict__ out) {
  const int T = 2048, HS = 64, CC = 1024;
  const int KSTR = 72;  // padded LDS stride for K / V^T tiles
  __shared__ bf16_t Ks[64 * KSTR];   // [kv][d]
  __shared__ bf16_t Vt[64 * KSTR];   // [d][kv]  (transposed V)
  __shared__ bf16_t Ps[8][16 * 64];  // per-wave P slab [q][kv]

  const int tid  = threadIdx.x;
  const int lane = tid & 31;
  const int w    = tid >> 5;
  const int g    = lane >> 4, n = lane & 15;
  const int bh   = blockIdx.y;          // b*H + h
  const int q0   = blockIdx.x * 128;    // query tile base
  const int qr0  = q0 + w * 16;         // this wave's first query row

  // Q fragments for d=0..31 and d=32..63, held in registers for the whole loop.
  const bf16_t* qbase = Q + ((size_t)bh * T + qr0) * HS;
  v16bf qf0 = load_a_frag(qbase, HS, lane);
  v16bf qf1 = load_a_frag(qbase + 32, HS, lane);

  v8f oacc[4];
#pragma unroll
  for (int dt = 0; dt < 4; ++dt) oacc[dt] = (v8f){};
  float M[8], L[8];
#pragma unroll
  for (int r = 0; r < 8; ++r) { M[r] = -3.0e38f; L[r] = 0.0f; }

  const int nkv  = (q0 + 128) >> 6;     // causal: kv tiles of 64 up to diag
  const int srow = tid >> 2, sseg = tid & 3;
  const unsigned ksoff = lds_off_of(&Ks[srow * KSTR + sseg * 16]);
  const float scale = 0.125f;           // hs^-0.5

  for (int it = 0; it < nkv; ++it) {
    const int kv0 = it * 64;
    __syncthreads();
    // Stage K tile [kv][d] via async DMA; V tile transposed [d][kv] manually.
    {
      const bf16_t* kgp = Kg + ((size_t)bh * T + kv0 + srow) * HS + sseg * 16;
      async_ld_b128(ksoff,      kgp);
      async_ld_b128(ksoff + 16, kgp + 8);
      const bf16_t* vgp = Vg + ((size_t)bh * T + kv0 + srow) * HS + sseg * 16;
      v8bf va = *(const v8bf*)(vgp);
      v8bf vb = *(const v8bf*)(vgp + 8);
#pragma unroll
      for (int i = 0; i < 8; ++i) {
        Vt[(sseg * 16 + i) * KSTR + srow]     = va[i];
        Vt[(sseg * 16 + 8 + i) * KSTR + srow] = vb[i];
      }
    }
    wait_async0();
    __syncthreads();

    // S = Q * K^T  (4 n-tiles of 16 keys, 2 k-steps over d)
    v8f sacc[4];
#pragma unroll
    for (int nt = 0; nt < 4; ++nt) {
      sacc[nt] = (v8f){};
      v16bf kf0 = load_b_frag(&Ks[(nt * 16) * KSTR], KSTR, lane);
      v16bf kf1 = load_b_frag(&Ks[(nt * 16) * KSTR + 32], KSTR, lane);
      sacc[nt] = WMMA_BF16(qf0, kf0, sacc[nt]);
      sacc[nt] = WMMA_BF16(qf1, kf1, sacc[nt]);
    }

    // Online softmax (rows live in a 16-lane group; vgpr r -> row g*8+r).
    float tmax[8];
#pragma unroll
    for (int r = 0; r < 8; ++r) tmax[r] = -3.0e38f;
#pragma unroll
    for (int nt = 0; nt < 4; ++nt)
#pragma unroll
      for (int r = 0; r < 8; ++r) {
        int kj = kv0 + nt * 16 + n;
        int qi = qr0 + g * 8 + r;
        float sv = sacc[nt][r] * scale;
        sv = (kj <= qi) ? sv : -3.0e38f;
        sacc[nt][r] = sv;
        tmax[r] = fmaxf(tmax[r], sv);
      }
#pragma unroll
    for (int r = 0; r < 8; ++r) {
      tmax[r] = fmaxf(tmax[r], __shfl_xor(tmax[r], 1));
      tmax[r] = fmaxf(tmax[r], __shfl_xor(tmax[r], 2));
      tmax[r] = fmaxf(tmax[r], __shfl_xor(tmax[r], 4));
      tmax[r] = fmaxf(tmax[r], __shfl_xor(tmax[r], 8));
    }
    float corr[8];
#pragma unroll
    for (int r = 0; r < 8; ++r) {
      float mn = fmaxf(M[r], tmax[r]);
      corr[r] = __expf(M[r] - mn);
      M[r] = mn;
    }
    float lsum[8];
#pragma unroll
    for (int r = 0; r < 8; ++r) lsum[r] = 0.0f;
#pragma unroll
    for (int nt = 0; nt < 4; ++nt)
#pragma unroll
      for (int r = 0; r < 8; ++r) {
        float p = __expf(sacc[nt][r] - M[r]);
        sacc[nt][r] = p;
        lsum[r] += p;
      }
#pragma unroll
    for (int r = 0; r < 8; ++r) {
      lsum[r] += __shfl_xor(lsum[r], 1);
      lsum[r] += __shfl_xor(lsum[r], 2);
      lsum[r] += __shfl_xor(lsum[r], 4);
      lsum[r] += __shfl_xor(lsum[r], 8);
      L[r] = L[r] * corr[r] + lsum[r];
    }
#pragma unroll
    for (int dt = 0; dt < 4; ++dt)
#pragma unroll
      for (int r = 0; r < 8; ++r) oacc[dt][r] *= corr[r];

    // P: C-layout -> A-layout via per-wave LDS slab (in-order per wave).
    bf16_t* pw = &Ps[w][0];
#pragma unroll
    for (int nt = 0; nt < 4; ++nt)
#pragma unroll
      for (int r = 0; r < 8; ++r)
        pw[(g * 8 + r) * 64 + nt * 16 + n] = (bf16_t)sacc[nt][r];
    asm volatile("s_wait_dscnt 0" ::: "memory");
    v16bf pf0 = load_a_frag(pw, 64, lane);
    v16bf pf1 = load_a_frag(pw + 32, 64, lane);

    // O += P * V  (B^T = Vt[d][kv])
#pragma unroll
    for (int dt = 0; dt < 4; ++dt) {
      v16bf vf0 = load_b_frag(&Vt[(dt * 16) * KSTR], KSTR, lane);
      v16bf vf1 = load_b_frag(&Vt[(dt * 16) * KSTR + 32], KSTR, lane);
      oacc[dt] = WMMA_BF16(pf0, vf0, oacc[dt]);
      oacc[dt] = WMMA_BF16(pf1, vf1, oacc[dt]);
    }
  }

  // Normalize and write out[b, t, h*64 + d] (fp32, coalesced over lanes).
  const int b = bh >> 4, h = bh & 15;
#pragma unroll
  for (int r = 0; r < 8; ++r) {
    float inv = 1.0f / L[r];
    int t = qr0 + g * 8 + r;
    float* op = out + ((size_t)b * T + t) * CC + h * 64 + n;
#pragma unroll
    for (int dt = 0; dt < 4; ++dt) op[dt * 16] = oacc[dt][r] * inv;
  }
}

// ---------------------------------------------------------------- launch ----
extern "C" void kernel_launch(void* const* d_in, const int* in_sizes, int n_in,
                              void* d_out, int out_size, void* d_ws, size_t ws_size,
                              hipStream_t stream) {
  const float* x    = (const float*)d_in[0];  // [4,2048,1024]
  const float* Wm   = (const float*)d_in[1];  // [3072,1024]
  const float* bias = (const float*)d_in[2];  // [3072]
  float* out        = (float*)d_out;          // [4,2048,1024]

  const int NX = 4 * 2048 * 1024;   // 8388608
  const int NW = 3072 * 1024;       // 3145728
  const size_t QKV = (size_t)4 * 16 * 2048 * 64;  // 8388608 elems each

  char* ws = (char*)d_ws;
  bf16_t* Xb = (bf16_t*)(ws);
  bf16_t* Wb = (bf16_t*)(ws + (size_t)NX * 2);
  bf16_t* Qb = (bf16_t*)(ws + (size_t)NX * 2 + (size_t)NW * 2);
  bf16_t* Kb = Qb + QKV;
  bf16_t* Vb = Kb + QKV;

  cvt_bf16_kernel<<<4096, 256, 0, stream>>>(x, Xb, NX);
  cvt_bf16_kernel<<<4096, 256, 0, stream>>>(Wm, Wb, NW);

  qkv_gemm_kernel<<<dim3(3072 / 128, 8192 / 128), 256, 0, stream>>>(
      Xb, Wb, bias, Qb, Kb, Vb);

  attn_kernel<<<dim3(2048 / 128, 4 * 16), 256, 0, stream>>>(Qb, Kb, Vb, out);
}
